// LSTMCell_17257178595687
// MI455X (gfx1250) — matrix operand order repaired and checked by reading, not compiled, not run on hardware
//
#include <hip/hip_runtime.h>
#include <hip/hip_bf16.h>
#include <math.h>

typedef __attribute__((ext_vector_type(16))) __bf16 v16bf;
typedef __attribute__((ext_vector_type(8)))  __bf16 v8bf;
typedef __attribute__((ext_vector_type(4)))  __bf16 v4bf;
typedef __attribute__((ext_vector_type(8)))  float  v8f;
typedef __attribute__((ext_vector_type(4)))  float  v4f;

#define BATCH   16384
#define IN_SZ   1024
#define H_SZ    1024
#define D_TOT   2048          // IN_SZ + H_SZ
#define BM      128           // batch rows per workgroup
#define BJ      32            // hidden columns per workgroup (per gate)
#define KT      32            // K per WMMA step (bf16 16x16x32)
#define NKT     (D_TOT / KT)  // 64
#define LDSW    40            // padded LDS row stride in bf16 elems (80 B, 16B aligned)

__global__ __launch_bounds__(256)
void lstm_cell_wmma_bf16(const float* __restrict__ x,
                         const float* __restrict__ cin,
                         const float* __restrict__ hin,
                         const float* __restrict__ Wf, const float* __restrict__ bfp,
                         const float* __restrict__ Wi, const float* __restrict__ bip,
                         const float* __restrict__ Wc, const float* __restrict__ bcp,
                         const float* __restrict__ Wo, const float* __restrict__ bop,
                         float* __restrict__ out)
{
    __shared__ __align__(16) __bf16 sA[2][BM * LDSW];   // xh tile: 128 rows x 32 k
    __shared__ __align__(16) __bf16 sW[2][BM * LDSW];   // W tile: 4 gates x 32 n x 32 k

    const int tid   = threadIdx.x;
    const int wave  = tid >> 5;     // 8 wave32s
    const int lane  = tid & 31;
    const int lrow  = lane & 15;
    const int lhalf = lane >> 4;

    const int jBase   = blockIdx.x * BJ;   // hidden-column tile (same cols in all 4 gates)
    const int rowBase = blockIdx.y * BM;   // batch-row tile

    // wave tile: 2 M-tiles (32 rows) x 4 gate tiles at column sub-tile cc (16 cols)
    const int mwave = (wave >> 1) * 32;    // 0,32,64,96
    const int cc    = wave & 1;            // 0 or 1 -> columns cc*16 .. cc*16+15

    // ---- global -> register prefetch mapping: each thread moves 16 A + 16 W elems / step
    const int r    = tid >> 1;         // 0..127 : A row / W n-row within tile
    const int k0   = (tid & 1) * 16;   // 0 or 16 inside the 32-wide k slice
    const int gate = r >> 5;           // which gate's weights this thread stages
    const int nIn  = r & 31;
    const float* wmat = (gate == 0) ? Wf : (gate == 1) ? Wi : (gate == 2) ? Wc : Wo;
    const float* wrow = wmat + (size_t)(jBase + nIn) * D_TOT;

    v4f pa[4], pw[4];
    auto gload = [&](int kt) {
        const int gk = kt * KT;                      // 0..2047, 32-aligned -> never straddles x|h
        const float* arow; int col;
        if (gk < IN_SZ) { arow = x   + (size_t)(rowBase + r) * IN_SZ; col = gk + k0; }
        else            { arow = hin + (size_t)(rowBase + r) * H_SZ;  col = gk - IN_SZ + k0; }
#pragma unroll
        for (int q = 0; q < 4; ++q) pa[q] = *(const v4f*)(arow + col + q * 4);
        const float* wp = wrow + gk + k0;
#pragma unroll
        for (int q = 0; q < 4; ++q) pw[q] = *(const v4f*)(wp + q * 4);
    };
    auto lstore = [&](int buf) {
        __bf16* da = &sA[buf][r * LDSW + k0];
        __bf16* dw = &sW[buf][r * LDSW + k0];
#pragma unroll
        for (int q = 0; q < 4; ++q) *(v4bf*)(da + q * 4) = __builtin_convertvector(pa[q], v4bf);
#pragma unroll
        for (int q = 0; q < 4; ++q) *(v4bf*)(dw + q * 4) = __builtin_convertvector(pw[q], v4bf);
    };

    // accumulators: acc[g*2 + mi] = 16x16 f32 tile (rows mwave+mi*16.., gate g, cols cc*16..)
    v8f acc[8];
    const v8f vzero = {0.f, 0.f, 0.f, 0.f, 0.f, 0.f, 0.f, 0.f};
#pragma unroll
    for (int t = 0; t < 8; ++t) acc[t] = vzero;

    int buf = 0;
    gload(0);
    lstore(0);

    for (int kt = 0; kt < NKT; ++kt) {
        __syncthreads();
        if (kt + 1 < NKT) gload(kt + 1);   // hide global latency under WMMA burst

        // ---- fragment loads: 2 A frags + 4 B frags = 12 ds_load_b128, all batched
        v16bf afrag[2];
#pragma unroll
        for (int mi = 0; mi < 2; ++mi) {
            const __bf16* pa0 = &sA[buf][(mwave + mi * 16 + lrow) * LDSW + lhalf * 8];
            v8bf alo = *(const v8bf*)(pa0);        // K = lhalf*8 .. +7
            v8bf ahi = *(const v8bf*)(pa0 + 16);   // K = 16 + lhalf*8 .. +7
            afrag[mi] = __builtin_shufflevector(alo, ahi,
                0,1,2,3,4,5,6,7,8,9,10,11,12,13,14,15);
        }
        v16bf bfrag[4];
#pragma unroll
        for (int g = 0; g < 4; ++g) {
            const __bf16* pb0 = &sW[buf][(g * 32 + cc * 16 + lrow) * LDSW + lhalf * 8];
            v8bf blo = *(const v8bf*)(pb0);        // K = lhalf*8 .. +7
            v8bf bhi = *(const v8bf*)(pb0 + 16);   // K = 16 + lhalf*8 .. +7
            bfrag[g] = __builtin_shufflevector(blo, bhi,
                0,1,2,3,4,5,6,7,8,9,10,11,12,13,14,15);
        }

        // pin loads before the WMMA burst so RA keeps them in distinct regs and
        // the wait pass can use partial dscnt counts
        __builtin_amdgcn_sched_barrier(0);

#pragma unroll
        for (int g = 0; g < 4; ++g) {
#pragma unroll
            for (int mi = 0; mi < 2; ++mi) {
                acc[g * 2 + mi] = __builtin_amdgcn_wmma_f32_16x16x32_bf16(
                    false, afrag[mi], false, bfrag[g], (short)0, acc[g * 2 + mi],
                    false, false);
            }
        }

        if (kt + 1 < NKT) lstore(buf ^ 1);
        buf ^= 1;
    }

    // ---- fused epilogue: bias + sigmoid/tanh + cell update, fp32
    const size_t CELLS = (size_t)BATCH * H_SZ;
    const int j = jBase + cc * 16 + lrow;
    const float bfv = bfp[j], biv = bip[j], bcv = bcp[j], bov = bop[j];
#pragma unroll
    for (int mi = 0; mi < 2; ++mi) {
#pragma unroll
        for (int v = 0; v < 8; ++v) {
            const int row = rowBase + mwave + mi * 16 + v + lhalf * 8;  // C-layout
            const float gf = acc[0 * 2 + mi][v] + bfv;
            const float gi = acc[1 * 2 + mi][v] + biv;
            const float gc = acc[2 * 2 + mi][v] + bcv;
            const float go = acc[3 * 2 + mi][v] + bov;
            const float ft = 1.f / (1.f + __expf(-gf));
            const float it = 1.f / (1.f + __expf(-gi));
            const float ct = tanhf(gc);
            const float ot = 1.f / (1.f + __expf(-go));
            const float cold = cin[(size_t)row * H_SZ + j];
            const float cn = ft * cold + it * ct;
            const float hn = ot * tanhf(cn);
            out[(size_t)row * H_SZ + j]         = cn;
            out[CELLS + (size_t)row * H_SZ + j] = hn;
        }
    }
}

extern "C" void kernel_launch(void* const* d_in, const int* in_sizes, int n_in,
                              void* d_out, int out_size, void* d_ws, size_t ws_size,
                              hipStream_t stream) {
    (void)in_sizes; (void)n_in; (void)out_size; (void)d_ws; (void)ws_size;
    const float* x  = (const float*)d_in[0];
    const float* c  = (const float*)d_in[1];
    const float* h  = (const float*)d_in[2];
    const float* Wf = (const float*)d_in[3];
    const float* bf = (const float*)d_in[4];
    const float* Wi = (const float*)d_in[5];
    const float* bi = (const float*)d_in[6];
    const float* Wc = (const float*)d_in[7];
    const float* bc = (const float*)d_in[8];
    const float* Wo = (const float*)d_in[9];
    const float* bo = (const float*)d_in[10];
    float* out = (float*)d_out;

    dim3 grid(H_SZ / BJ, BATCH / BM);   // (32, 128) = 4096 workgroups
    lstm_cell_wmma_bf16<<<grid, dim3(256), 0, stream>>>(
        x, c, h, Wf, bf, Wi, bi, Wc, bc, Wo, bo, out);
}
